// DA_RNN_2585570312333
// MI455X (gfx1250) — compile-verified, matrix operand order checked by reading, not run on hardware
//
#include <hip/hip_runtime.h>
#include <hip/hip_bf16.h>

// ---------------- CDNA5 WMMA types ----------------
typedef __bf16 bf16_t;
typedef __attribute__((ext_vector_type(16))) __bf16 v16bf;
typedef __attribute__((ext_vector_type(8)))  float  v8f;

// ---------------- problem constants ----------------
static constexpr int BN  = 2048;   // batch
static constexpr int TSN = 9;      // T-1 timesteps
static constexpr int HN  = 512;    // hidden
static constexpr int H2  = 2 * HN; // 1024
static constexpr int IN1 = 16;
static constexpr int IN2 = 15;

// ============================================================
// device helpers
// ============================================================
__device__ __forceinline__ float sigf(float x) { return 1.0f / (1.0f + __expf(-x)); }

__device__ __forceinline__ float wave_sum(float v) {
#pragma unroll
  for (int off = 16; off > 0; off >>= 1) v += __shfl_down(v, off, 32);
  return v;
}

// Fragment pointer for a 16-row bf16 tile: lane L covers row (L&15),
// 16 contiguous K values at 16*(L>>4).  Same consistent K-permutation of the
// HW layout for A and B -> dot products pair correctly.
__device__ __forceinline__ const bf16_t* frag_ptr(const bf16_t* base, int ld, int lane) {
  return base + (long)(lane & 15) * ld + ((lane >> 4) << 4);
}
__device__ __forceinline__ v16bf frag_ld(const bf16_t* p, int k0) {
  return *(const v16bf*)(p + k0);  // 32B aligned: ld mult of 16, k0 mult of 32
}

// ============================================================
// K0: build x_y (B,9,16) and mm padded (B,9,16)
// ============================================================
__global__ void darnn_prep(const float* __restrict__ X, const float* __restrict__ yprev,
                           float* __restrict__ xy, float* __restrict__ mmp) {
  long idx = (long)blockIdx.x * blockDim.x + threadIdx.x;
  if (idx >= (long)BN * TSN * 16) return;
  int i = idx & 15;
  long bt = idx >> 4;
  float y = yprev[bt];
  float xv = (i < 15) ? X[bt * 15 + i] : 0.0f;
  xy[idx]  = (i < 15) ? xv : y;
  mmp[idx] = (i < 15) ? xv * y : 0.0f;
}

// K0b: base1[b,i] = sum_t xy[b,t,i]*Wf[t] + b0 ; base2 likewise from mm
__global__ void darnn_base(const float* __restrict__ xy, const float* __restrict__ mmp,
                           const float* __restrict__ encW, const float* __restrict__ b0,
                           float* __restrict__ base1, float* __restrict__ base2) {
  int idx = blockIdx.x * blockDim.x + threadIdx.x;
  if (idx >= BN * 16) return;
  int b = idx >> 4, i = idx & 15;
  const float* Wf = encW + 2 * HN;
  float a1 = b0[0], a2 = b0[0];
#pragma unroll
  for (int t = 0; t < TSN; ++t) {
    a1 += xy[((long)b * TSN + t) * 16 + i] * Wf[t];
    a2 += mmp[((long)b * TSN + t) * 16 + i] * Wf[t];
  }
  base1[idx] = a1;
  base2[idx] = a2;
}

// K1: pack f32 weights -> bf16 row-major (dst cols contiguous)
__global__ void darnn_pack(const float* __restrict__ src, bf16_t* __restrict__ dst,
                           int cols, int sld, long total) {
  long i = (long)blockIdx.x * blockDim.x + threadIdx.x;
  if (i >= total) return;
  long r = i / cols;
  int  c = (int)(i - r * cols);
  dst[i] = (bf16_t)src[r * sld + c];
}

// ============================================================
// K2: encoder input attention (one wave per batch row)
// ============================================================
__global__ void darnn_enc_attn(const float* __restrict__ h, const float* __restrict__ s,
                               const float* __restrict__ base, const float* __restrict__ encW,
                               const float* __restrict__ xin, float* __restrict__ xt,
                               int t, int IN) {
  int wid  = (blockIdx.x * blockDim.x + threadIdx.x) >> 5;
  int lane = threadIdx.x & 31;
  if (wid >= BN) return;
  int b = wid;
  float acc = 0.0f;
  for (int k = lane; k < HN; k += 32)
    acc += h[(long)b * HN + k] * encW[k] + s[(long)b * HN + k] * encW[HN + k];
  acc = wave_sum(acc);
  acc = __shfl(acc, 0, 32);
  if (lane < 16) {
    float v = (lane < IN) ? base[b * 16 + lane] + acc : -1e30f;
    float mx = v;
#pragma unroll
    for (int off = 1; off < 16; off <<= 1) mx = fmaxf(mx, __shfl_xor(mx, off, 32));
    float e = (lane < IN) ? __expf(v - mx) : 0.0f;
    float sm = e;
#pragma unroll
    for (int off = 1; off < 16; off <<= 1) sm += __shfl_xor(sm, off, 32);
    float a = e / sm;
    xt[b * 16 + lane] = (lane < IN) ? a * xin[((long)b * TSN + t) * 16 + lane] : 0.0f;
  }
}

// ============================================================
// K3/K7: fused WMMA GEMM + LSTM cell update, register-blocked.
// One wave owns MT m-tiles x one 16-wide j slice, with 4 gate accumulators
// per m-tile.  Fragments are loaded just-in-time: LLVM's pressure-aware
// scheduler pipelines/clauses the b128 loads across the WMMA block without
// forcing double-buffer liveness (manual double-buffering spills).
// ============================================================
template <int MT>
__global__ void darnn_lstm_step(const bf16_t* __restrict__ hin, const bf16_t* __restrict__ Whh,
                                const float* __restrict__ xt, const float* __restrict__ Wih,
                                const float* __restrict__ bih, const float* __restrict__ bhh,
                                float* __restrict__ cbuf, float* __restrict__ hout, int ldh,
                                float* __restrict__ hout2, int ldh2,
                                bf16_t* __restrict__ hbf,
                                bf16_t* __restrict__ hbf2, int ldh2b,
                                bf16_t* __restrict__ cbf,
                                int K, int Hd, int IN, int mode) {
  int wid  = (blockIdx.x * blockDim.x + threadIdx.x) >> 5;
  int lane = threadIdx.x & 31;
  int jt   = Hd >> 4;
  int mtg  = wid / jt;
  int jtt  = wid - mtg * jt;

  v8f acc[MT][4];
#pragma unroll
  for (int mi = 0; mi < MT; ++mi)
#pragma unroll
    for (int g = 0; g < 4; ++g) acc[mi][g] = v8f{};

  const bf16_t* Ab[MT];
#pragma unroll
  for (int mi = 0; mi < MT; ++mi)
    Ab[mi] = frag_ptr(hin + (long)(mtg * MT + mi) * 16 * K, K, lane);
  const bf16_t* Bb[4];
#pragma unroll
  for (int g = 0; g < 4; ++g)
    Bb[g] = frag_ptr(Whh + (long)(g * Hd + jtt * 16) * K, K, lane);

  for (int k0 = 0; k0 < K; k0 += 32) {
    v16bf a[MT];
#pragma unroll
    for (int mi = 0; mi < MT; ++mi) a[mi] = frag_ld(Ab[mi], k0);
#pragma unroll
    for (int g = 0; g < 4; ++g) {
      v16bf b = frag_ld(Bb[g], k0);
#pragma unroll
      for (int mi = 0; mi < MT; ++mi)
        acc[mi][g] = __builtin_amdgcn_wmma_f32_16x16x32_bf16(false, a[mi], false, b,
                                                             (short)0, acc[mi][g], false, false);
    }
  }

  int col = jtt * 16 + (lane & 15);
#pragma unroll
  for (int mi = 0; mi < MT; ++mi) {
    int row0 = (mtg * MT + mi) * 16 + ((lane >> 4) << 3);
#pragma unroll
    for (int r = 0; r < 8; ++r) {
      int row = row0 + r;
      float xr[16];
      if (mode == 0) {
        const float* xp = xt + (long)row * 16;
#pragma unroll
        for (int k = 0; k < 16; ++k) xr[k] = xp[k];
      }
      float g4[4];
#pragma unroll
      for (int g = 0; g < 4; ++g) {
        int n = g * Hd + col;
        float xw = 0.0f;
        if (mode == 0) {
          const float* wr = Wih + (long)n * IN;
          for (int k = 0; k < IN; ++k) xw += xr[k] * wr[k];
        } else {
          xw = xt[row] * Wih[n];
        }
        g4[g] = acc[mi][g][r] + bih[n] + bhh[n] + xw;
      }
      float iv = sigf(g4[0]), fv = sigf(g4[1]);
      float gv = tanhf(g4[2]), ov = sigf(g4[3]);
      long ci = (long)row * Hd + col;
      float cn = fv * cbuf[ci] + iv * gv;
      cbuf[ci] = cn;
      if (cbf) cbf[ci] = (bf16_t)cn;
      float hn = ov * tanhf(cn);
      hout[(long)row * ldh + col] = hn;
      hbf[ci] = (bf16_t)hn;   // bf16 state feeds next step's WMMA directly
      if (hout2) hout2[(long)row * ldh2 + col] = hn;
      if (hbf2)  hbf2[(long)row * ldh2b + col] = (bf16_t)hn;
    }
  }
}

// ============================================================
// K4/K5: generic WMMA GEMM, register-blocked MT m-tiles per wave.
// C[m,n] = sum_k A1[m,k]*W1[n,k] (+ A2@W2) (+ bias[n]); all operands bf16.
// ============================================================
template <int MT>
__global__ void darnn_gemm(const bf16_t* __restrict__ A1, const bf16_t* __restrict__ W1,
                           const bf16_t* __restrict__ A2, const bf16_t* __restrict__ W2,
                           const float* __restrict__ bias, float* __restrict__ C,
                           int N, int K, int lda, int ldc) {
  int wid  = (blockIdx.x * blockDim.x + threadIdx.x) >> 5;
  int lane = threadIdx.x & 31;
  int nt   = N >> 4;
  int mtg  = wid / nt;
  int ntt  = wid - mtg * nt;

  v8f acc[MT];
#pragma unroll
  for (int mi = 0; mi < MT; ++mi) acc[mi] = v8f{};

  {
    const bf16_t* Ab[MT];
#pragma unroll
    for (int mi = 0; mi < MT; ++mi)
      Ab[mi] = frag_ptr(A1 + (long)(mtg * MT + mi) * 16 * lda, lda, lane);
    const bf16_t* Wb = frag_ptr(W1 + (long)ntt * 16 * K, K, lane);
    for (int k0 = 0; k0 < K; k0 += 32) {
      v16bf b = frag_ld(Wb, k0);
#pragma unroll
      for (int mi = 0; mi < MT; ++mi)
        acc[mi] = __builtin_amdgcn_wmma_f32_16x16x32_bf16(false, frag_ld(Ab[mi], k0), false, b,
                                                          (short)0, acc[mi], false, false);
    }
  }
  if (A2) {
    const bf16_t* Ab[MT];
#pragma unroll
    for (int mi = 0; mi < MT; ++mi)
      Ab[mi] = frag_ptr(A2 + (long)(mtg * MT + mi) * 16 * lda, lda, lane);
    const bf16_t* Wb = frag_ptr(W2 + (long)ntt * 16 * K, K, lane);
    for (int k0 = 0; k0 < K; k0 += 32) {
      v16bf b = frag_ld(Wb, k0);
#pragma unroll
      for (int mi = 0; mi < MT; ++mi)
        acc[mi] = __builtin_amdgcn_wmma_f32_16x16x32_bf16(false, frag_ld(Ab[mi], k0), false, b,
                                                          (short)0, acc[mi], false, false);
    }
  }

  int col = ntt * 16 + (lane & 15);
  float bv = bias ? bias[col] : 0.0f;
#pragma unroll
  for (int mi = 0; mi < MT; ++mi) {
    int row0 = (mtg * MT + mi) * 16 + ((lane >> 4) << 3);
#pragma unroll
    for (int r = 0; r < 8; ++r)
      C[(long)(row0 + r) * ldc + col] = acc[mi][r] + bv;
  }
}

// ============================================================
// K6: decoder temporal attention + context + y_tilde (one block per batch row)
// ============================================================
__global__ void darnn_dec_attn(const float* __restrict__ hx, const float* __restrict__ dcw,
                               const float* __restrict__ Xenc, const float* __restrict__ w2,
                               const float* __restrict__ b2, const float* __restrict__ fcW,
                               const float* __restrict__ fcb, const float* __restrict__ yprev,
                               int t, float* __restrict__ ctx, float* __restrict__ yt) {
  int b = blockIdx.x, tid = threadIdx.x;  // 512 threads, 16 waves
  int lane = tid & 31, wv = tid >> 5;
  __shared__ float sred[16 * TSN];
  __shared__ float sc[TSN];
  __shared__ float sbeta[TSN];
  __shared__ float sy[16];

  float w2v = w2[tid];
  float dc  = dcw[(long)b * HN + tid];
  float loc[TSN];
#pragma unroll
  for (int k = 0; k < TSN; ++k)
    loc[k] = (hx[((long)b * TSN + k) * HN + tid] + dc) * w2v;
#pragma unroll
  for (int off = 16; off > 0; off >>= 1)
#pragma unroll
    for (int k = 0; k < TSN; ++k) loc[k] += __shfl_down(loc[k], off, 32);
  if (lane == 0)
#pragma unroll
    for (int k = 0; k < TSN; ++k) sred[wv * TSN + k] = loc[k];
  __syncthreads();
  if (tid < TSN) {
    float s = b2[0];
    for (int w = 0; w < 16; ++w) s += sred[w * TSN + tid];
    sc[tid] = s;
  }
  __syncthreads();
  if (tid == 0) {
    float mx = sc[0];
#pragma unroll
    for (int k = 1; k < TSN; ++k) mx = fmaxf(mx, sc[k]);
    float e[TSN], sum = 0.0f;
#pragma unroll
    for (int k = 0; k < TSN; ++k) { e[k] = __expf(sc[k] - mx); sum += e[k]; }
#pragma unroll
    for (int k = 0; k < TSN; ++k) sbeta[k] = e[k] / sum;
  }
  __syncthreads();
  float c0 = 0.0f, c1 = 0.0f;
#pragma unroll
  for (int k = 0; k < TSN; ++k) {
    float bt = sbeta[k];
    const float* xe = Xenc + ((long)b * TSN + k) * H2;
    c0 += bt * xe[tid];
    c1 += bt * xe[tid + HN];
  }
  ctx[(long)b * H2 + tid]      = c0;
  ctx[(long)b * H2 + tid + HN] = c1;
  float p = c0 * fcW[tid] + c1 * fcW[tid + HN];
  p = wave_sum(p);
  if (lane == 0) sy[wv] = p;
  __syncthreads();
  if (tid == 0) {
    float s = 0.0f;
    for (int w = 0; w < 16; ++w) s += sy[w];
    yt[b] = s + fcW[H2] * yprev[(long)b * TSN + t] + fcb[0];
  }
}

// ============================================================
// K8: y_pred[b] = fc_final_W . [d, context] + b
// ============================================================
__global__ void darnn_final(const float* __restrict__ d, const float* __restrict__ ctx,
                            const float* __restrict__ W, const float* __restrict__ bb,
                            float* __restrict__ out) {
  int b = blockIdx.x, tid = threadIdx.x, lane = tid & 31, wv = tid >> 5;  // 256 thr
  __shared__ float sred[8];
  float acc = 0.0f;
  for (int k = tid; k < 2 * H2; k += 256)
    acc += W[k] * (k < H2 ? d[(long)b * H2 + k] : ctx[(long)b * H2 + k - H2]);
  acc = wave_sum(acc);
  if (lane == 0) sred[wv] = acc;
  __syncthreads();
  if (tid == 0) {
    float s = 0.0f;
    for (int w = 0; w < 8; ++w) s += sred[w];
    out[b] = s + bb[0];
  }
}

// ============================================================
// host launcher
// ============================================================
extern "C" void kernel_launch(void* const* d_in, const int* in_sizes, int n_in,
                              void* d_out, int out_size, void* d_ws, size_t ws_size,
                              hipStream_t stream) {
  const float* X          = (const float*)d_in[0];
  const float* yprev      = (const float*)d_in[1];
  const float* enc_attn_W = (const float*)d_in[2];
  const float* enc_attn_b = (const float*)d_in[3];
  const float* eWih  = (const float*)d_in[4];
  const float* eWhh  = (const float*)d_in[5];
  const float* ebih  = (const float*)d_in[6];
  const float* ebhh  = (const float*)d_in[7];
  const float* e1Wih = (const float*)d_in[8];
  const float* e1Whh = (const float*)d_in[9];
  const float* e1bih = (const float*)d_in[10];
  const float* e1bhh = (const float*)d_in[11];
  const float* a1W   = (const float*)d_in[12];
  const float* a1b   = (const float*)d_in[13];
  const float* a2W   = (const float*)d_in[14];
  const float* a2b   = (const float*)d_in[15];
  const float* dWih  = (const float*)d_in[16];
  const float* dWhh  = (const float*)d_in[17];
  const float* dbih  = (const float*)d_in[18];
  const float* dbhh  = (const float*)d_in[19];
  const float* fcW   = (const float*)d_in[20];
  const float* fcb   = (const float*)d_in[21];
  const float* ffW   = (const float*)d_in[22];
  const float* ffb   = (const float*)d_in[23];
  float* out = (float*)d_out;

  size_t off = 0;
  auto alloc = [&](size_t bytes) -> void* {
    off = (off + 255) & ~(size_t)255;
    void* p = (char*)d_ws + off;
    off += bytes;
    return p;
  };
  float* xy    = (float*)alloc((size_t)BN * TSN * 16 * 4);
  float* mmp   = (float*)alloc((size_t)BN * TSN * 16 * 4);
  float* base1 = (float*)alloc((size_t)BN * 16 * 4);
  float* base2 = (float*)alloc((size_t)BN * 16 * 4);
  float* xt1   = (float*)alloc((size_t)BN * 16 * 4);
  float* xt2   = (float*)alloc((size_t)BN * 16 * 4);
  float* h1a   = (float*)alloc((size_t)BN * HN * 4);
  float* h1b   = (float*)alloc((size_t)BN * HN * 4);
  float* s1    = (float*)alloc((size_t)BN * HN * 4);
  float* h2a   = (float*)alloc((size_t)BN * HN * 4);
  float* h2b   = (float*)alloc((size_t)BN * HN * 4);
  float* s2    = (float*)alloc((size_t)BN * HN * 4);
  float* Xenc  = (float*)alloc((size_t)BN * TSN * H2 * 4);
  float* hx    = (float*)alloc((size_t)BN * TSN * HN * 4);
  float* dcw   = (float*)alloc((size_t)BN * HN * 4);
  float* da    = (float*)alloc((size_t)BN * H2 * 4);
  float* db    = (float*)alloc((size_t)BN * H2 * 4);
  float* cdec  = (float*)alloc((size_t)BN * H2 * 4);
  float* ctx   = (float*)alloc((size_t)BN * H2 * 4);
  float* yt    = (float*)alloc((size_t)BN * 4);
  // bf16 mirrors of dynamic state (written by LSTM epilogues, feed WMMA)
  bf16_t* h1bfa = (bf16_t*)alloc((size_t)BN * HN * 2);
  bf16_t* h1bfb = (bf16_t*)alloc((size_t)BN * HN * 2);
  bf16_t* h2bfa = (bf16_t*)alloc((size_t)BN * HN * 2);
  bf16_t* h2bfb = (bf16_t*)alloc((size_t)BN * HN * 2);
  bf16_t* XencB = (bf16_t*)alloc((size_t)BN * TSN * H2 * 2);
  bf16_t* dbfa  = (bf16_t*)alloc((size_t)BN * H2 * 2);
  bf16_t* dbfb  = (bf16_t*)alloc((size_t)BN * H2 * 2);
  bf16_t* cbfD  = (bf16_t*)alloc((size_t)BN * H2 * 2);
  // bf16 packed weights
  bf16_t* eWhhB  = (bf16_t*)alloc((size_t)4 * HN * HN * 2);
  bf16_t* e1WhhB = (bf16_t*)alloc((size_t)4 * HN * HN * 2);
  bf16_t* dWhhB  = (bf16_t*)alloc((size_t)4 * H2 * H2 * 2);
  bf16_t* W1dB   = (bf16_t*)alloc((size_t)HN * H2 * 2);
  bf16_t* W1cB   = (bf16_t*)alloc((size_t)HN * H2 * 2);
  bf16_t* W1xB   = (bf16_t*)alloc((size_t)HN * H2 * 2);
  (void)ws_size; (void)n_in; (void)in_sizes; (void)out_size;

  // zero recurrent state (every call: deterministic)
  hipMemsetAsync(h1a, 0, (size_t)BN * HN * 4, stream);
  hipMemsetAsync(s1,  0, (size_t)BN * HN * 4, stream);
  hipMemsetAsync(h2a, 0, (size_t)BN * HN * 4, stream);
  hipMemsetAsync(s2,  0, (size_t)BN * HN * 4, stream);
  hipMemsetAsync(da,  0, (size_t)BN * H2 * 4, stream);
  hipMemsetAsync(cdec,0, (size_t)BN * H2 * 4, stream);
  hipMemsetAsync(h1bfa, 0, (size_t)BN * HN * 2, stream);
  hipMemsetAsync(h2bfa, 0, (size_t)BN * HN * 2, stream);
  hipMemsetAsync(dbfa,  0, (size_t)BN * H2 * 2, stream);
  hipMemsetAsync(cbfD,  0, (size_t)BN * H2 * 2, stream);

  // prep + bases
  darnn_prep<<<(BN * TSN * 16 + 255) / 256, 256, 0, stream>>>(X, yprev, xy, mmp);
  darnn_base<<<(BN * 16 + 255) / 256, 256, 0, stream>>>(xy, mmp, enc_attn_W, enc_attn_b, base1, base2);

  // pack weights to bf16
  {
    long n;
    n = (long)4 * HN * HN;
    darnn_pack<<<(int)((n + 255) / 256), 256, 0, stream>>>(eWhh,  eWhhB,  HN, HN, n);
    darnn_pack<<<(int)((n + 255) / 256), 256, 0, stream>>>(e1Whh, e1WhhB, HN, HN, n);
    n = (long)4 * H2 * H2;
    darnn_pack<<<(int)((n + 255) / 256), 256, 0, stream>>>(dWhh, dWhhB, H2, H2, n);
    n = (long)HN * H2;
    darnn_pack<<<(int)((n + 255) / 256), 256, 0, stream>>>(a1W,          W1dB, H2, 3 * H2, n);
    darnn_pack<<<(int)((n + 255) / 256), 256, 0, stream>>>(a1W + H2,     W1cB, H2, 3 * H2, n);
    darnn_pack<<<(int)((n + 255) / 256), 256, 0, stream>>>(a1W + 2 * H2, W1xB, H2, 3 * H2, n);
  }

  // -------- encoder: 9 sequential steps, two attention-LSTMs --------
  const int attnBlocks = BN / 8;
  const int encLstmBlocks = ((BN / 16 / 2) * (HN / 16)) * 32 / 256;  // MT=2 -> 256
  float* h1c = h1a; float* h1n = h1b;  bf16_t* h1bc = h1bfa; bf16_t* h1bn = h1bfb;
  float* h2c = h2a; float* h2n = h2b;  bf16_t* h2bc = h2bfa; bf16_t* h2bn = h2bfb;
  for (int t = 0; t < TSN; ++t) {
    darnn_enc_attn<<<attnBlocks, 256, 0, stream>>>(h1c, s1, base1, enc_attn_W, xy,  xt1, t, IN1);
    darnn_enc_attn<<<attnBlocks, 256, 0, stream>>>(h2c, s2, base2, enc_attn_W, mmp, xt2, t, IN2);
    darnn_lstm_step<2><<<encLstmBlocks, 256, 0, stream>>>(
        h1bc, eWhhB, xt1, eWih, ebih, ebhh, s1, h1n, HN,
        Xenc + (long)t * H2, TSN * H2,
        h1bn, XencB + (long)t * H2, TSN * H2, nullptr, HN, HN, IN1, 0);
    darnn_lstm_step<2><<<encLstmBlocks, 256, 0, stream>>>(
        h2bc, e1WhhB, xt2, e1Wih, e1bih, e1bhh, s2, h2n, HN,
        Xenc + (long)t * H2 + HN, TSN * H2,
        h2bn, XencB + (long)t * H2 + HN, TSN * H2, nullptr, HN, HN, IN2, 0);
    float* tf; bf16_t* tb;
    tf = h1c; h1c = h1n; h1n = tf;   tb = h1bc; h1bc = h1bn; h1bn = tb;
    tf = h2c; h2c = h2n; h2n = tf;   tb = h2bc; h2bc = h2bn; h2bn = tb;
  }

  // -------- hx = Xenc @ W1x^T  (18432 x 512, K=1024), MT=4 --------
  {
    int waves = ((BN * TSN / 16) / 4) * (HN / 16);  // 288*32 = 9216
    darnn_gemm<4><<<waves * 32 / 256, 256, 0, stream>>>(XencB, W1xB, nullptr, nullptr, nullptr,
                                                        hx, HN, H2, H2, HN);
  }

  // -------- decoder: 9 sequential steps --------
  const int dcwBlocks     = ((BN / 16 / 2) * (HN / 16)) * 32 / 256;  // MT=2 -> 256
  const int decLstmBlocks = ((BN / 16 / 2) * (H2 / 16)) * 32 / 256;  // MT=2 -> 512
  float* dc = da; float* dn = db;  bf16_t* dbc = dbfa; bf16_t* dbn = dbfb;
  for (int t = 0; t < TSN; ++t) {
    darnn_gemm<2><<<dcwBlocks, 256, 0, stream>>>(dbc, W1dB, cbfD, W1cB, a1b, dcw, HN, H2, H2, HN);
    darnn_dec_attn<<<BN, 512, 0, stream>>>(hx, dcw, Xenc, a2W, a2b, fcW, fcb, yprev, t, ctx, yt);
    darnn_lstm_step<2><<<decLstmBlocks, 256, 0, stream>>>(
        dbc, dWhhB, yt, dWih, dbih, dbhh, cdec, dn, H2,
        nullptr, 0, dbn, nullptr, 0, cbfD, H2, H2, 1, 1);
    float* tf = dc; dc = dn; dn = tf;
    bf16_t* tb = dbc; dbc = dbn; dbn = tb;
  }

  // -------- final projection --------
  darnn_final<<<BN, 256, 0, stream>>>(dc, ctx, ffW, ffb, out);
}
